// LSTMTagger_16449724743985
// MI455X (gfx1250) — compile-verified, matrix operand order checked by reading, not run on hardware
//
#include <hip/hip_runtime.h>
#include <hip/hip_bf16.h>

// ---------------------------------------------------------------------------
// LSTM tagger for MI455X (gfx1250, wave32, WMMA, async-to-LDS staging).
// Sizes: S=4096, L=16, E=512, H=512, 4H=2048, V=100000, T=64, A=50.
//   * GEMMs in f16 (values ~0.05) with f32 WMMA accumulation.
//   * GEMM staging: GLOBAL_LOAD_ASYNC_TO_LDS_B128 + double-buffered LDS,
//     synchronized with s_wait_asynccnt (ASYNCcnt) + workgroup barriers.
//   * Char LSTM: 16 steps of fused gates-GEMM (x_t*Wih^T + h*Whh^T) + cell.
//   * Word LSTM: batch-1 latency chain -> persistent 1024-thread workgroup.
// ---------------------------------------------------------------------------

#define S_TOK 4096
#define L_CHR 16
#define E_DIM 512
#define H_DIM 512
#define G_DIM 2048   /* 4*H */
#define T_TAG 64
#define KTOT  512
#define LDS_STRIDE 40   /* halfs per LDS tile row (80 B, keeps b32 reads 4B aligned) */

typedef __attribute__((ext_vector_type(16))) _Float16 v16h;
typedef __attribute__((ext_vector_type(8)))  _Float16 v8h;
typedef __attribute__((ext_vector_type(2)))  _Float16 v2h;
typedef __attribute__((ext_vector_type(8)))  float    v8f;

// ---------------------------------------------------------------- helpers
__device__ __forceinline__ float sigf(float x) { return 1.0f / (1.0f + __expf(-x)); }

template <int N>
__device__ __forceinline__ void wait_async() {
#if __has_builtin(__builtin_amdgcn_s_wait_asynccnt)
  __builtin_amdgcn_s_wait_asynccnt(N);
#else
  asm volatile("s_wait_asynccnt %0" :: "i"(N) : "memory");
#endif
}

// One 16B global -> LDS async transfer per lane (ASYNCcnt-tracked, no VGPR data).
__device__ __forceinline__ void async_b128(unsigned lds_addr, const void* gaddr) {
  asm volatile("global_load_async_to_lds_b128 %0, %1, off"
               :: "v"(lds_addr), "v"(gaddr) : "memory");
}

// Build a 16x32 f16 fragment (A-layout per ISA 7.12.2; B mirrored with
// lane = column) from an LDS tile stored [row][LDS_STRIDE].
__device__ __forceinline__ v16h load_frag(const _Float16* tile, int row, int half) {
  v16h f;
  const _Float16* rp = tile + row * LDS_STRIDE;
#pragma unroll
  for (int p = 0; p < 8; ++p) {
    int kb = (p < 4) ? (p * 2 + half * 8) : (16 + (p - 4) * 2 + half * 8);
    v2h t = *(const v2h*)(rp + kb);
    f[2 * p]     = t[0];
    f[2 * p + 1] = t[1];
  }
  return f;
}

// Asynchronously stage a 128x32 f16 tile into LDS: 256 threads x 2 async
// 16B transfers. Rows >= rowLimit are clamped (so every wave issues exactly
// 2 instructions here); their LDS slots are zeroed post-wait by zero_oob().
__device__ __forceinline__ void stage_async(_Float16* dst, const _Float16* src,
                                            long rowBase, int kb, int tid,
                                            long rowLimit) {
#pragma unroll
  for (int it = 0; it < 2; ++it) {
    int c   = tid + it * 256;   // 0..511
    int r   = c >> 2;           // tile row 0..127
    int seg = c & 3;            // 16B segment within the 64B row chunk
    long row = rowBase + r;
    if (row >= rowLimit) row = rowLimit - 1;
    const void* g = (const void*)(src + row * KTOT + kb + seg * 8);
    unsigned lds = (unsigned)(uintptr_t)(dst + r * LDS_STRIDE + seg * 8);
    async_b128(lds, g);
  }
}

__device__ __forceinline__ void zero_oob(_Float16* dst, long rowBase,
                                         long rowLimit, int tid) {
  if (rowBase + 128 <= rowLimit) return;  // common fast path
#pragma unroll
  for (int it = 0; it < 2; ++it) {
    int c   = tid + it * 256;
    int r   = c >> 2;
    int seg = c & 3;
    if (rowBase + r >= rowLimit) {
      v8h z = {};
      *(v8h*)(dst + r * LDS_STRIDE + seg * 8) = z;
    }
  }
}

// ------------------------------------------------------------ WMMA GEMM
// C[M,N] = A1[M,512]*W1[N,512]^T (+ A2[M,512]*W2[N,512]^T) + bias1 (+bias2)
// 256 threads = 8 waves; block tile 128x128; wave tile 32x64 (2x4 WMMA accs).
// K chunks of 32, double-buffered LDS fed by async-to-LDS copies.
__global__ __launch_bounds__(256) void gemm_f16_wmma(
    const _Float16* __restrict__ A1, const _Float16* __restrict__ W1,
    const _Float16* __restrict__ A2, const _Float16* __restrict__ W2,
    const float* __restrict__ bias1, const float* __restrict__ bias2,
    float* __restrict__ C, int M, int N) {
  __shared__ _Float16 As[2][128 * LDS_STRIDE];
  __shared__ _Float16 Bs[2][128 * LDS_STRIDE];

  const int tid  = threadIdx.x;
  const int lane = tid & 31;
  const int wv   = tid >> 5;
  const int half = lane >> 4;
  const int lrow = lane & 15;
  const int mBase = (wv & 3) * 32;   // wave M offset in block tile
  const int nBase = (wv >> 2) * 64;  // wave N offset in block tile
  const long mBlock = (long)blockIdx.y * 128;
  const long nBlock = (long)blockIdx.x * 128;

  v8f acc[2][4];
#pragma unroll
  for (int i = 0; i < 2; ++i)
#pragma unroll
    for (int j = 0; j < 4; ++j)
      acc[i][j] = (v8f){0.f, 0.f, 0.f, 0.f, 0.f, 0.f, 0.f, 0.f};

  const int nChunks = (A2 != nullptr) ? 32 : 16;  // two fused K passes

  auto issue = [&](int ch, int b) {
    const _Float16* A = (ch < 16) ? A1 : A2;
    const _Float16* W = (ch < 16) ? W1 : W2;
    int kb = (ch & 15) * 32;
    stage_async(&As[b][0], A, mBlock, kb, tid, 0x7fffffffL);
    stage_async(&Bs[b][0], W, nBlock, kb, tid, (long)N);
  };

  issue(0, 0);
  int buf = 0;
#pragma unroll 1
  for (int ch = 0; ch < nChunks; ++ch) {
    if (ch + 1 < nChunks) {
      issue(ch + 1, buf ^ 1);   // prefetch next chunk into other buffer
      wait_async<4>();          // this chunk's 4 async ops have landed
    } else {
      wait_async<0>();
    }
    zero_oob(&Bs[buf][0], nBlock, (long)N, tid);
    __syncthreads();

    v16h af[2], bf[4];
#pragma unroll
    for (int i = 0; i < 2; ++i) af[i] = load_frag(&As[buf][0], mBase + i * 16 + lrow, half);
#pragma unroll
    for (int j = 0; j < 4; ++j) bf[j] = load_frag(&Bs[buf][0], nBase + j * 16 + lrow, half);

#pragma unroll
    for (int i = 0; i < 2; ++i)
#pragma unroll
      for (int j = 0; j < 4; ++j)
        acc[i][j] = __builtin_amdgcn_wmma_f32_16x16x32_f16(
            false, af[i], false, bf[j], (short)0, acc[i][j], false, false);

    __syncthreads();
    buf ^= 1;
  }

  // Epilogue: C/D layout -> lane 0-15: N=lane, M=r; lane 16-31: N=lane-16, M=r+8
#pragma unroll
  for (int i = 0; i < 2; ++i) {
#pragma unroll
    for (int j = 0; j < 4; ++j) {
      long n = nBlock + nBase + j * 16 + lrow;
      if (n < N) {
        float b = bias1 ? bias1[n] : 0.f;
        if (bias2) b += bias2[n];
#pragma unroll
        for (int r = 0; r < 8; ++r) {
          long m = mBlock + mBase + i * 16 + r + half * 8;
          C[m * (long)N + n] = acc[i][j][r] + b;
        }
      }
    }
  }
}

// ------------------------------------------------------- small kernels
__global__ void cvt_f32_to_f16(const float* __restrict__ src,
                               _Float16* __restrict__ dst, int n) {
  int i = blockIdx.x * blockDim.x + threadIdx.x;
  if (i < n) dst[i] = (_Float16)src[i];
}

__global__ void gather_word_f16(const int* __restrict__ sent_in,
                                const float* __restrict__ word_emb,
                                _Float16* __restrict__ xw) {
  int i = blockIdx.x * blockDim.x + threadIdx.x;  // over S*E
  int s = i >> 9, e = i & 511;
  xw[i] = (_Float16)word_emb[(long)sent_in[s] * E_DIM + e];
}

__global__ void gather_char_f16(const int* __restrict__ char_in,
                                const float* __restrict__ char_emb,
                                _Float16* __restrict__ xc) {
  long i = (long)blockIdx.x * blockDim.x + threadIdx.x;  // over L*S*E
  int e = (int)(i & 511);
  int s = (int)((i >> 9) & (S_TOK - 1));
  int t = (int)(i >> 21);
  xc[i] = (_Float16)char_emb[(long)char_in[s * L_CHR + t] * E_DIM + e];
}

__global__ void init_char_state(float* __restrict__ h, float* __restrict__ c,
                                float* __restrict__ osum, _Float16* __restrict__ hh) {
  int i = blockIdx.x * blockDim.x + threadIdx.x;  // over S*H
  h[i] = 0.f; c[i] = 0.f; osum[i] = 0.f; hh[i] = (_Float16)0.f;
}

// Elementwise LSTM cell for the char path (gate order i, f, g, o; carry
// frozen on masked steps, out_sum only accumulates valid steps).
__global__ void lstm_cell_char(const float* __restrict__ gates,
                               const int* __restrict__ lengths, int t,
                               float* __restrict__ h, _Float16* __restrict__ hh,
                               float* __restrict__ c, float* __restrict__ osum) {
  int i = blockIdx.x * blockDim.x + threadIdx.x;  // over S*H
  int s = i >> 9, j = i & 511;
  const float* g = gates + (long)s * G_DIM;
  float ig = sigf(g[j]);
  float fg = sigf(g[512 + j]);
  float gg = tanhf(g[1024 + j]);
  float og = sigf(g[1536 + j]);
  float c_new = fg * c[i] + ig * gg;
  float h_new = og * tanhf(c_new);
  bool valid = t < lengths[s];
  if (valid) {
    c[i] = c_new;
    h[i] = h_new;
    osum[i] += h_new;
    hh[i] = (_Float16)h_new;
  }
  // if masked: h/c/hh keep previous values, osum unchanged
}

// Word LSTM: batch 1, 4096 sequential steps. Single persistent workgroup.
// gates(2048) = gx[t] + Whh(2048x512,f16)*h + bhh ; cell in LDS.
__global__ __launch_bounds__(1024) void word_lstm(
    const float* __restrict__ gx,      // [4096, 2048], includes bih
    const _Float16* __restrict__ whh,  // [2048, 512] f16
    const float* __restrict__ bhh,     // [2048]
    _Float16* __restrict__ hs) {       // [4096, 512] f16 out
  __shared__ float hS[H_DIM];
  __shared__ float cS[H_DIM];
  __shared__ float gS[G_DIM];
  const int tid = threadIdx.x;
  if (tid < H_DIM) { hS[tid] = 0.f; cS[tid] = 0.f; }
  __syncthreads();

  for (int t = 0; t < S_TOK; ++t) {
#pragma unroll
    for (int rr = 0; rr < 2; ++rr) {
      int r = tid + rr * 1024;
      const v8h* wr = (const v8h*)(whh + (long)r * H_DIM);
      float acc = 0.f;
#pragma unroll 4
      for (int q = 0; q < H_DIM / 8; ++q) {
        v8h v = wr[q];
#pragma unroll
        for (int u = 0; u < 8; ++u) acc += (float)v[u] * hS[q * 8 + u];
      }
      if (t + 1 < S_TOK)  // hide L2 latency of next step's gx row
        __builtin_prefetch(&gx[(long)(t + 1) * G_DIM + r], 0, 1);
      gS[r] = gx[(long)t * G_DIM + r] + acc + bhh[r];
    }
    __syncthreads();
    if (tid < H_DIM) {
      float ig = sigf(gS[tid]);
      float fg = sigf(gS[512 + tid]);
      float gg = tanhf(gS[1024 + tid]);
      float og = sigf(gS[1536 + tid]);
      float cn = fg * cS[tid] + ig * gg;
      float hn = og * tanhf(cn);
      cS[tid] = cn; hS[tid] = hn;
      hs[(long)t * H_DIM + tid] = (_Float16)hn;
    }
    __syncthreads();
  }
}

// out = log_softmax(word_scores + char_scores) over T=64.
// One wave32 per row, 2 columns per lane, shuffle reductions.
__global__ __launch_bounds__(256) void out_logsoftmax(
    const float* __restrict__ ws, const float* __restrict__ cs,
    float* __restrict__ out) {
  int wave = threadIdx.x >> 5;
  int lane = threadIdx.x & 31;
  long row = (long)blockIdx.x * 8 + wave;
  const float* w = ws + row * T_TAG;
  const float* c = cs + row * T_TAG;
  float z0 = w[lane] + c[lane];
  float z1 = w[32 + lane] + c[32 + lane];
  float m = fmaxf(z0, z1);
#pragma unroll
  for (int off = 16; off > 0; off >>= 1) m = fmaxf(m, __shfl_xor(m, off, 32));
  float e = __expf(z0 - m) + __expf(z1 - m);
#pragma unroll
  for (int off = 16; off > 0; off >>= 1) e += __shfl_xor(e, off, 32);
  float lse = m + __logf(e);
  out[row * T_TAG + lane]      = z0 - lse;
  out[row * T_TAG + 32 + lane] = z1 - lse;
}

// ---------------------------------------------------------------- launch
extern "C" void kernel_launch(void* const* d_in, const int* in_sizes, int n_in,
                              void* d_out, int out_size, void* d_ws, size_t ws_size,
                              hipStream_t stream) {
  const int*   sent_in  = (const int*)  d_in[0];
  const int*   char_in  = (const int*)  d_in[1];
  const int*   lengths  = (const int*)  d_in[2];
  const float* word_emb = (const float*)d_in[3];
  const float* char_emb = (const float*)d_in[4];
  const float* Wih_w    = (const float*)d_in[5];
  const float* Whh_w    = (const float*)d_in[6];
  const float* bih_w    = (const float*)d_in[7];
  const float* bhh_w    = (const float*)d_in[8];
  const float* Wih_c    = (const float*)d_in[9];
  const float* Whh_c    = (const float*)d_in[10];
  const float* bih_c    = (const float*)d_in[11];
  const float* bhh_c    = (const float*)d_in[12];
  const float* Ww2t     = (const float*)d_in[13];
  const float* bw2t     = (const float*)d_in[14];
  const float* Wc2t     = (const float*)d_in[15];
  const float* bc2t     = (const float*)d_in[16];

  char* ws = (char*)d_ws;
  size_t off = 0;
  auto take = [&](size_t bytes) -> void* {
    void* p = ws + off;
    off += (bytes + 255) & ~(size_t)255;
    return p;
  };

  _Float16* wih_w_h = (_Float16*)take((size_t)G_DIM * E_DIM * 2);
  _Float16* whh_w_h = (_Float16*)take((size_t)G_DIM * H_DIM * 2);
  _Float16* wih_c_h = (_Float16*)take((size_t)G_DIM * E_DIM * 2);
  _Float16* whh_c_h = (_Float16*)take((size_t)G_DIM * H_DIM * 2);
  _Float16* ww2t_h  = (_Float16*)take((size_t)T_TAG * H_DIM * 2);
  _Float16* wc2t_h  = (_Float16*)take((size_t)T_TAG * H_DIM * 2);
  _Float16* xw_h    = (_Float16*)take((size_t)S_TOK * E_DIM * 2);
  _Float16* xc_h    = (_Float16*)take((size_t)L_CHR * S_TOK * E_DIM * 2);
  float*    gxw     = (float*)   take((size_t)S_TOK * G_DIM * 4);
  _Float16* hsw_h   = (_Float16*)take((size_t)S_TOK * H_DIM * 2);
  float*    gates_c = (float*)   take((size_t)S_TOK * G_DIM * 4);
  float*    h_c     = (float*)   take((size_t)S_TOK * H_DIM * 4);
  _Float16* h_c_h   = (_Float16*)take((size_t)S_TOK * H_DIM * 2);
  float*    c_c     = (float*)   take((size_t)S_TOK * H_DIM * 4);
  float*    osum    = (float*)   take((size_t)S_TOK * H_DIM * 4);
  _Float16* osum_h  = (_Float16*)take((size_t)S_TOK * H_DIM * 2);
  float*    wscore  = (float*)   take((size_t)S_TOK * T_TAG * 4);
  float*    cscore  = (float*)   take((size_t)S_TOK * T_TAG * 4);
  (void)ws_size; (void)n_in; (void)in_sizes; (void)out_size;

  const int TPB = 256;

  // 1) weight conversions f32 -> f16
  cvt_f32_to_f16<<<(G_DIM * E_DIM + TPB - 1) / TPB, TPB, 0, stream>>>(Wih_w, wih_w_h, G_DIM * E_DIM);
  cvt_f32_to_f16<<<(G_DIM * H_DIM + TPB - 1) / TPB, TPB, 0, stream>>>(Whh_w, whh_w_h, G_DIM * H_DIM);
  cvt_f32_to_f16<<<(G_DIM * E_DIM + TPB - 1) / TPB, TPB, 0, stream>>>(Wih_c, wih_c_h, G_DIM * E_DIM);
  cvt_f32_to_f16<<<(G_DIM * H_DIM + TPB - 1) / TPB, TPB, 0, stream>>>(Whh_c, whh_c_h, G_DIM * H_DIM);
  cvt_f32_to_f16<<<(T_TAG * H_DIM + TPB - 1) / TPB, TPB, 0, stream>>>(Ww2t, ww2t_h, T_TAG * H_DIM);
  cvt_f32_to_f16<<<(T_TAG * H_DIM + TPB - 1) / TPB, TPB, 0, stream>>>(Wc2t, wc2t_h, T_TAG * H_DIM);

  // 2) embedding gathers (-> f16 GEMM operands)
  gather_word_f16<<<(S_TOK * E_DIM) / TPB, TPB, 0, stream>>>(sent_in, word_emb, xw_h);
  gather_char_f16<<<(L_CHR * S_TOK * E_DIM) / TPB, TPB, 0, stream>>>(char_in, char_emb, xc_h);

  // 3) word-path input gates: gxw = xw * Wih_w^T + bih_w
  {
    dim3 grid(G_DIM / 128, S_TOK / 128);
    gemm_f16_wmma<<<grid, TPB, 0, stream>>>(xw_h, wih_w_h, nullptr, nullptr,
                                            bih_w, nullptr, gxw, S_TOK, G_DIM);
  }

  // 4) word LSTM recurrence (sequential, persistent workgroup)
  word_lstm<<<1, 1024, 0, stream>>>(gxw, whh_w_h, bhh_w, hsw_h);

  // 5) char LSTM: init state then 16 steps of fused gates-GEMM + cell
  init_char_state<<<(S_TOK * H_DIM) / TPB, TPB, 0, stream>>>(h_c, c_c, osum, h_c_h);
  {
    dim3 grid(G_DIM / 128, S_TOK / 128);
    for (int t = 0; t < L_CHR; ++t) {
      gemm_f16_wmma<<<grid, TPB, 0, stream>>>(
          xc_h + (size_t)t * S_TOK * E_DIM, wih_c_h, h_c_h, whh_c_h,
          bih_c, bhh_c, gates_c, S_TOK, G_DIM);
      lstm_cell_char<<<(S_TOK * H_DIM) / TPB, TPB, 0, stream>>>(
          gates_c, lengths, t, h_c, h_c_h, c_c, osum);
    }
  }

  // 6) projections to tag space (N=64 edge handled by clamp + zero_oob + store guard)
  cvt_f32_to_f16<<<(S_TOK * H_DIM) / TPB, TPB, 0, stream>>>(osum, osum_h, S_TOK * H_DIM);
  {
    dim3 grid(1, S_TOK / 128);
    gemm_f16_wmma<<<grid, TPB, 0, stream>>>(hsw_h, ww2t_h, nullptr, nullptr,
                                            bw2t, nullptr, wscore, S_TOK, T_TAG);
    gemm_f16_wmma<<<grid, TPB, 0, stream>>>(osum_h, wc2t_h, nullptr, nullptr,
                                            bc2t, nullptr, cscore, S_TOK, T_TAG);
  }

  // 7) add + log_softmax -> d_out (f32, [4096, 64])
  out_logsoftmax<<<S_TOK / 8, TPB, 0, stream>>>(wscore, cscore, (float*)d_out);
}
